// SelfAttention_867583394222
// MI455X (gfx1250) — matrix operand order repaired and checked by reading, compiled, not run on hardware
//
#include <hip/hip_runtime.h>

typedef __bf16 bf16_t;
typedef __attribute__((ext_vector_type(16))) __bf16 v16bf;
typedef __attribute__((ext_vector_type(8)))  __bf16 v8bf;
typedef __attribute__((ext_vector_type(2)))  __bf16 v2bf;
typedef __attribute__((ext_vector_type(8)))  float  v8f;

#define DIM 4096
#define NHEADS 32
#define NKV 8
#define HD 128
#define BATCH 2
#define SEQ 2048
#define MTOT (BATCH*SEQ)   // 4096

// cheap fp32 -> bf16 (round-half-up; ties differ from RNE, negligible)
__device__ __forceinline__ unsigned short f2bf(float f) {
  return (unsigned short)((__float_as_uint(f) + 0x8000u) >> 16);
}

// packed fp32x2 -> bf16x2: one v_perm_b32 after the round-bias adds
__device__ __forceinline__ unsigned int f2bf2(float lo, float hi) {
#if __has_builtin(__builtin_amdgcn_cvt_pk_bf16_f32)
  v2bf r = __builtin_amdgcn_cvt_pk_bf16_f32(lo, hi);
  return __builtin_bit_cast(unsigned int, r);
#else
  unsigned int ul = __float_as_uint(lo) + 0x8000u;
  unsigned int uh = __float_as_uint(hi) + 0x8000u;
  // bytes: [uh.b3, uh.b2, ul.b3, ul.b2]
  return __builtin_amdgcn_perm(uh, ul, 0x07060302u);
#endif
}

__device__ __forceinline__ v16bf comb(v8bf lo, v8bf hi) {
  return __builtin_shufflevector(lo, hi, 0,1,2,3,4,5,6,7,8,9,10,11,12,13,14,15);
}

__device__ __forceinline__ v8f wmma_bf16(v16bf a, v16bf b, v8f c) {
  return __builtin_amdgcn_wmma_f32_16x16x32_bf16(false, a, false, b, (short)0, c, false, false);
}

// ---------------------------------------------------------------------------
// NT GEMM: C[M,N] = sum_k A[m,k]*B[n,k].  A: MxK fp32, B: NxK fp32, C: MxN fp32.
// Block tile 128x128, BK=64 (two WMMA k-steps per stage). 256 thr = 8 waves,
// wave tile 32(M) x 64(N).  LDS rows padded to 72 halfs (36 dwords) so the
// 16 lanes of a fragment read land in disjoint 4-bank groups.
// ---------------------------------------------------------------------------
#define LDA 72
__global__ __launch_bounds__(256, 2)
void gemm_nt_bf16(const float* __restrict__ A, const float* __restrict__ Bm,
                  float* __restrict__ C, int M, int N, int K) {
  __shared__ unsigned short As[128 * LDA];
  __shared__ unsigned short Bs[128 * LDA];
  const int t    = threadIdx.x;
  const int lane = t & 31;
  const int wave = t >> 5;
  const int hlf  = lane >> 4;
  const int l16  = lane & 15;
  const int wm   = wave & 3;
  const int wn   = wave >> 2;
  const int mBlk = blockIdx.y * 128;
  const int nBlk = blockIdx.x * 128;

  v8f acc[2][4];
#pragma unroll
  for (int i = 0; i < 2; ++i)
#pragma unroll
    for (int j = 0; j < 4; ++j) acc[i][j] = {};

  for (int kt = 0; kt < K; kt += 64) {
    // ---- stage A & B tiles (fp32 -> bf16, v_perm pack) ----
#pragma unroll
    for (int i = 0; i < 8; ++i) {
      int idx = t + i * 256;                 // 0..2047
      int row = idx >> 4;                    // 16 float4 per 64-wide row
      int c4  = (idx & 15) << 2;
      float4 va = *(const float4*)(A + (size_t)(mBlk + row) * K + kt + c4);
      uint2* da = (uint2*)&As[row * LDA + c4];
      *da = make_uint2(f2bf2(va.x, va.y), f2bf2(va.z, va.w));
      float4 vb = *(const float4*)(Bm + (size_t)(nBlk + row) * K + kt + c4);
      uint2* db = (uint2*)&Bs[row * LDA + c4];
      *db = make_uint2(f2bf2(vb.x, vb.y), f2bf2(vb.z, vb.w));
    }
    __syncthreads();

#pragma unroll
    for (int kk = 0; kk < 2; ++kk) {
      v16bf af[2], bfr[4];
#pragma unroll
      for (int i = 0; i < 2; ++i) {
        const unsigned short* p = &As[(wm * 32 + i * 16 + l16) * LDA + kk * 32];
        af[i] = comb(*(const v8bf*)(p + hlf * 16), *(const v8bf*)(p + hlf * 16 + 8));
      }
#pragma unroll
      for (int j = 0; j < 4; ++j) {
        const unsigned short* p = &Bs[(wn * 64 + j * 16 + l16) * LDA + kk * 32 + hlf * 16];
        bfr[j] = comb(*(const v8bf*)p, *(const v8bf*)(p + 8));
      }
#pragma unroll
      for (int i = 0; i < 2; ++i)
#pragma unroll
        for (int j = 0; j < 4; ++j) acc[i][j] = wmma_bf16(af[i], bfr[j], acc[i][j]);
    }
    __syncthreads();
  }

#pragma unroll
  for (int i = 0; i < 2; ++i)
#pragma unroll
    for (int j = 0; j < 4; ++j)
#pragma unroll
      for (int r = 0; r < 8; ++r) {
        int row = mBlk + wm * 32 + i * 16 + r + 8 * hlf;
        int col = nBlk + wn * 64 + j * 16 + l16;
        C[(size_t)row * N + col] = acc[i][j][r];
      }
}

// ---------------------------------------------------------------------------
// RoPE (fp32 math) + optional scale + convert to bf16; one u32 store per pair.
// The softmax 1/sqrt(HD) is folded into q here.
// ---------------------------------------------------------------------------
__global__ void rope_to_bf16(const float* __restrict__ X,
                             const float* __restrict__ fc,
                             const float* __restrict__ fs,
                             unsigned int* __restrict__ Y, int nheads,
                             float oscale) {
  size_t i = (size_t)blockIdx.x * 256 + threadIdx.x;
  size_t total = (size_t)BATCH * SEQ * nheads * (HD / 2);
  if (i >= total) return;
  int j = i & 63;
  size_t r = i >> 6;                     // (b*S + s)*nheads + h
  int s = (int)((r / nheads) % SEQ);
  float c = fc[s * 64 + j], sn = fs[s * 64 + j];
  size_t base = r * HD + 2 * j;
  float xr = X[base], xi = X[base + 1];
  Y[r * 64 + j] = f2bf2((xr * c - xi * sn) * oscale, (xr * sn + xi * c) * oscale);
}

// Convert V fp32 (B,S,KV,128) -> bf16 transposed (B,KV,128,S)
__global__ void v_transpose_bf16(const float* __restrict__ X,
                                 unsigned short* __restrict__ Y) {
  size_t i = (size_t)blockIdx.x * 256 + threadIdx.x;   // < B*S*KV*128
  int d  = i & 127;
  int kv = (i >> 7) & 7;
  int s  = (i >> 10) & 2047;
  int b  = (int)(i >> 21);
  Y[(((size_t)b * NKV + kv) * HD + d) * SEQ + s] = f2bf(X[i]);
}

// ---------------------------------------------------------------------------
// Flash attention (non-causal).  Q bf16 (B,S,32,128) pre-scaled by 1/sqrt(HD),
// K bf16 (B,S,8,128), Vt bf16 (B,8,128,S).  Out fp32 (B,S,4096) head-major.
// Grid (S/128, NHEADS, B); all 8 waves share (b,h) -> K/Vt tiles staged
// cooperatively into padded LDS once per 32-key step.
// ---------------------------------------------------------------------------
#define LDK 136   // 32 x 128-half K tile rows padded to 136 halfs
#define LDV 40    // 128 x 32-half Vt tile rows padded to 40 halfs
__global__ __launch_bounds__(256, 1)
void flash_attn(const unsigned short* __restrict__ Q,
                const unsigned short* __restrict__ Kc,
                const unsigned short* __restrict__ Vt,
                float* __restrict__ Out) {
  __shared__ unsigned short Ks[32 * LDK];
  __shared__ unsigned short Vs[128 * LDV];
  __shared__ unsigned short Pl[8][16][LDV];
  const int t    = threadIdx.x;
  const int lane = t & 31;
  const int wave = t >> 5;
  const int hlf  = lane >> 4;
  const int l16  = lane & 15;
  const int h    = blockIdx.y;
  const int b    = blockIdx.z;
  const int kvh  = h >> 2;                   // n_rep = 4
  const int q0   = blockIdx.x * 128 + wave * 16;

  v16bf aq[4];
  {
    const unsigned short* qp = Q + ((size_t)(b * SEQ + q0 + l16) * NHEADS + h) * HD;
#pragma unroll
    for (int kc = 0; kc < 4; ++kc)
      aq[kc] = comb(*(const v8bf*)(qp + kc * 32 + hlf * 16),
                    *(const v8bf*)(qp + kc * 32 + hlf * 16 + 8));
  }

  float m_i[8], l_i[8];
  v8f o[8];
#pragma unroll
  for (int r = 0; r < 8; ++r) { m_i[r] = -1e30f; l_i[r] = 0.f; }
#pragma unroll
  for (int j = 0; j < 8; ++j) o[j] = {};

  const unsigned short* kbase = Kc + (size_t)b * SEQ * NKV * HD + (size_t)kvh * HD;
  const unsigned short* vbase = Vt + ((size_t)b * NKV + kvh) * HD * SEQ;

  const int krow = t >> 3, kcol = (t & 7) * 16;   // K: 32 rows x 128 halfs
  const int vrow = t >> 1, vcol = (t & 1) * 16;   // Vt: 128 rows x 32 halfs

  for (int s0 = 0; s0 < SEQ; s0 += 32) {
    { // ---- cooperative stage of K (32x128) and Vt (128x32) tiles ----
      const uint4* ksrc = (const uint4*)(kbase + (size_t)(s0 + krow) * NKV * HD + kcol);
      uint4* kdst = (uint4*)&Ks[krow * LDK + kcol];
      kdst[0] = ksrc[0]; kdst[1] = ksrc[1];
      const uint4* vsrc = (const uint4*)(vbase + (size_t)vrow * SEQ + s0 + vcol);
      uint4* vdst = (uint4*)&Vs[vrow * LDV + vcol];
      vdst[0] = vsrc[0]; vdst[1] = vsrc[1];
    }
    __syncthreads();

    // ---- scores: 16x32 via 8 WMMAs from LDS (scale pre-folded into Q) ----
    v8f sc0 = {}, sc1 = {};
#pragma unroll
    for (int kc = 0; kc < 4; ++kc) {
      const unsigned short* kp0 = &Ks[l16 * LDK + kc * 32 + hlf * 16];
      const unsigned short* kp1 = &Ks[(16 + l16) * LDK + kc * 32 + hlf * 16];
      v16bf bk0 = comb(*(const v8bf*)kp0, *(const v8bf*)(kp0 + 8));
      v16bf bk1 = comb(*(const v8bf*)kp1, *(const v8bf*)(kp1 + 8));
      sc0 = wmma_bf16(aq[kc], bk0, sc0);
      sc1 = wmma_bf16(aq[kc], bk1, sc1);
    }

    // ---- online softmax (rows r+8*hlf, cols striped over 16 lanes) ----
    float alpha[8];
#pragma unroll
    for (int r = 0; r < 8; ++r) {
      float a0 = sc0[r], a1 = sc1[r];
      float mx = fmaxf(a0, a1);
#pragma unroll
      for (int off = 8; off >= 1; off >>= 1) mx = fmaxf(mx, __shfl_xor(mx, off, 32));
      float mn = fmaxf(m_i[r], mx);
      alpha[r] = __expf(m_i[r] - mn);
      m_i[r]   = mn;
      float p0 = __expf(a0 - mn), p1 = __expf(a1 - mn);
      float rs = p0 + p1;
#pragma unroll
      for (int off = 8; off >= 1; off >>= 1) rs += __shfl_xor(rs, off, 32);
      l_i[r] = l_i[r] * alpha[r] + rs;
      int row = r + 8 * hlf;
      Pl[wave][row][l16]      = f2bf(p0);
      Pl[wave][row][16 + l16] = f2bf(p1);
    }
#pragma unroll
    for (int j = 0; j < 8; ++j) {
      v8f tv = o[j];
#pragma unroll
      for (int r = 0; r < 8; ++r) tv[r] *= alpha[r];
      o[j] = tv;
    }

    asm volatile("s_wait_dscnt 0" ::: "memory");   // wave-local P writes -> reads

    const unsigned short* pr = &Pl[wave][l16][0];
    v16bf pa = comb(*(const v8bf*)(pr + hlf * 16), *(const v8bf*)(pr + hlf * 16 + 8));
#pragma unroll
    for (int j = 0; j < 8; ++j) {
      const unsigned short* vp = &Vs[(j * 16 + l16) * LDV + hlf * 16];
      v16bf bv = comb(*(const v8bf*)vp, *(const v8bf*)(vp + 8));
      o[j] = wmma_bf16(pa, bv, o[j]);
    }
    __syncthreads();
  }

#pragma unroll
  for (int j = 0; j < 8; ++j)
#pragma unroll
    for (int r = 0; r < 8; ++r) {
      int row = q0 + r + 8 * hlf;
      Out[(size_t)(b * SEQ + row) * DIM + h * HD + j * 16 + l16] = o[j][r] / l_i[r];
    }
}

// ---------------------------------------------------------------------------
extern "C" void kernel_launch(void* const* d_in, const int* in_sizes, int n_in,
                              void* d_out, int out_size, void* d_ws, size_t ws_size,
                              hipStream_t stream) {
  const float* x  = (const float*)d_in[0];
  const float* wq = (const float*)d_in[1];
  const float* wk = (const float*)d_in[2];
  const float* wv = (const float*)d_in[3];
  const float* wo = (const float*)d_in[4];
  const float* fc = (const float*)d_in[5];
  const float* fs = (const float*)d_in[6];
  // d_in[7..9]: cache_k, cache_v, start_pos (=0) -> not needed

  char* ws = (char*)d_ws;
  const size_t SZ_XQ = (size_t)MTOT * DIM * 4;        // 64 MB
  const size_t SZ_XK = (size_t)MTOT * NKV * HD * 4;   // 16 MB
  float* xq = (float*)ws;                              // reused as attn output
  float* xk = (float*)(ws + SZ_XQ);
  float* xv = (float*)(ws + SZ_XQ + SZ_XK);
  unsigned short* qbf = (unsigned short*)(ws + SZ_XQ + 2 * SZ_XK);
  unsigned short* kbf = (unsigned short*)(ws + SZ_XQ + 2 * SZ_XK + (size_t)MTOT * DIM * 2);
  unsigned short* vtb = (unsigned short*)(ws + SZ_XQ + 2 * SZ_XK + (size_t)MTOT * DIM * 2
                                             + (size_t)MTOT * NKV * HD * 2);

  dim3 blk(256);
  gemm_nt_bf16<<<dim3(DIM / 128, MTOT / 128), blk, 0, stream>>>(x, wq, xq, MTOT, DIM, DIM);
  gemm_nt_bf16<<<dim3(NKV * HD / 128, MTOT / 128), blk, 0, stream>>>(x, wk, xk, MTOT, NKV * HD, DIM);
  gemm_nt_bf16<<<dim3(NKV * HD / 128, MTOT / 128), blk, 0, stream>>>(x, wv, xv, MTOT, NKV * HD, DIM);
  rope_to_bf16<<<(BATCH * SEQ * NHEADS * 64) / 256, blk, 0, stream>>>(
      xq, fc, fs, (unsigned int*)qbf, NHEADS, 0.08838834764831845f /* 1/sqrt(128) */);
  rope_to_bf16<<<(BATCH * SEQ * NKV * 64) / 256, blk, 0, stream>>>(
      xk, fc, fs, (unsigned int*)kbf, NKV, 1.0f);
  v_transpose_bf16<<<(BATCH * SEQ * NKV * HD) / 256, blk, 0, stream>>>(xv, vtb);
  flash_attn<<<dim3(SEQ / 128, NHEADS, BATCH), blk, 0, stream>>>(qbf, kbf, vtb, xq);
  gemm_nt_bf16<<<dim3(DIM / 128, MTOT / 128), blk, 0, stream>>>(xq, wo, (float*)d_out, MTOT, DIM, DIM);
}
